// TransformerLayer_87497073754298
// MI455X (gfx1250) — matrix operand order repaired and checked by reading, not compiled, not run on hardware
//
#include <hip/hip_runtime.h>
#include <hip/hip_bf16.h>
#include <stdint.h>

// ---------------------------------------------------------------------------
// MI455X (gfx1250) transformer layer:
//   Q = Y W^T ; A = softmax(Q Q^T) ; Z = A Y ; out = relu(Z * (-0.5(Wf+Wf^T)) + B)
// GEMMs: v_wmma_f32_16x16x32_bf16, double-buffered LDS staging via
// global_load_async_to_lds_b128 (ASYNCcnt pipelined against WMMA).
// ---------------------------------------------------------------------------

typedef __attribute__((ext_vector_type(16))) __bf16 v16bf;
typedef __attribute__((ext_vector_type(8)))  float  v8f;

union FragBF { v16bf v; uint32_t u[8]; };
union AccF   { v8f v; float f[8]; };

__device__ __forceinline__ uint16_t f2bf(float f) {
    union { float f; uint32_t u; } c; c.f = f;
    uint32_t u = c.u;
    return (uint16_t)((u + 0x7FFFu + ((u >> 16) & 1u)) >> 16);  // RNE
}
__device__ __forceinline__ float bf2f(uint16_t h) {
    union { uint32_t u; float f; } c; c.u = ((uint32_t)h) << 16;
    return c.f;
}

// Low 32 bits of a generic shared-memory pointer == LDS byte address
__device__ __forceinline__ uint32_t lds_off(const void* p) {
    return (uint32_t)(uintptr_t)p;
}

// CDNA5 async global->LDS copy, 16 bytes per lane, tracked by ASYNCcnt.
__device__ __forceinline__ void async_copy_b128(uint32_t lds_byte, const void* gaddr) {
    asm volatile("global_load_async_to_lds_b128 %0, %1, off"
                 :: "v"(lds_byte), "v"(gaddr) : "memory");
}
template <int N>
__device__ __forceinline__ void wait_async_le() {
    asm volatile("s_wait_asynccnt %0" :: "i"(N) : "memory");
}

// A fragment (16x32 bf16, ISA 7.12.2): lane half h -> 16B at k=8h and k=16+8h
__device__ __forceinline__ void load_fragA(FragBF& f, const uint16_t* rowp, int h) {
    *(uint4*)&f.u[0] = *(const uint4*)(rowp + 8 * h);
    *(uint4*)&f.u[4] = *(const uint4*)(rowp + 16 + 8 * h);
}
// B fragment (32x16 bf16): lane = column, half h -> 16B at k=8h and k=16+8h
__device__ __forceinline__ void load_fragB(FragBF& f, const uint16_t* rowp, int h) {
    *(uint4*)&f.u[0] = *(const uint4*)(rowp + 8 * h);
    *(uint4*)&f.u[4] = *(const uint4*)(rowp + 16 + 8 * h);
}

// --------------------------- prep kernels ----------------------------------

__global__ void prep_y(const float* __restrict__ Y, uint16_t* __restrict__ Ybf,
                       uint16_t* __restrict__ Ytb, int nrow, int ncol) {
    int idx = blockIdx.x * blockDim.x + threadIdx.x;
    if (idx >= nrow * ncol) return;
    int i = idx / ncol, j = idx - i * ncol;
    uint16_t b = f2bf(Y[idx]);
    Ybf[idx] = b;
    Ytb[(size_t)j * nrow + i] = b;   // transposed copy for the A@Y GEMM
}

__global__ void prep_w(const float* __restrict__ W, uint16_t* __restrict__ Wbf, int total) {
    int idx = blockIdx.x * blockDim.x + threadIdx.x;
    if (idx >= total) return;
    Wbf[idx] = f2bf(W[idx]);
}

__global__ void prep_wf(const float* __restrict__ Wf, uint16_t* __restrict__ Wfe, int d) {
    int idx = blockIdx.x * blockDim.x + threadIdx.x;
    if (idx >= d * d) return;
    int i = idx / d, j = idx - i * d;
    Wfe[idx] = f2bf(-0.5f * (Wf[idx] + Wf[(size_t)j * d + i]));  // symmetric
}

// --------------------------- WMMA NT GEMM ----------------------------------
// C[M,N] = A[M,K] * Brows[N,K]^T   (bf16 operands, contiguous along K).
// Block: 256 threads = 8 waves, 128x128 tile, K-tile 64, double-buffered LDS.
// Per K-tile: async-copy next A/B 128x64 tiles into the other buffer while
// each wave computes 32x64 x K64 (16 WMMAs) from the current one.
// EPI==0: store bf16.  EPI==1: C fp32, add bias[col], relu.
template <int EPI>
__global__ __launch_bounds__(256)
void gemm_nt_wmma(const uint16_t* __restrict__ A, int lda,
                  const uint16_t* __restrict__ Bm, int ldb,
                  void* __restrict__ Cv, int ldc, int K,
                  const float* __restrict__ bias) {
    constexpr int KT = 64;                      // K-tile
    constexpr int LS = KT + 8;                  // padded LDS row (elements)
    __shared__ __align__(16) uint16_t As[2][128 * LS];
    __shared__ __align__(16) uint16_t Bs[2][128 * LS];

    const int tid  = threadIdx.x;
    const int lane = tid & 31;
    const int wid  = tid >> 5;                  // 0..7
    const int wm   = wid >> 1;                  // 0..3  -> 32-row strip
    const int wn   = wid & 1;                   // 0..1  -> 64-col strip
    const int h    = lane >> 4;
    const int l16  = lane & 15;

    const int row0 = blockIdx.y * 128;
    const int col0 = blockIdx.x * 128;

    // global->LDS: 128 rows x 128B = 1024 16B-chunks per tile; 4 per thread.
    // chunk id = tid + 256*i : row = (tid>>3) + 32*i, c = tid&7.
    const int r = tid >> 3, c = tid & 7;
    const uint16_t* gA[4];
    const uint16_t* gB[4];
    uint32_t lA[2][4], lB[2][4];
#pragma unroll
    for (int i = 0; i < 4; ++i) {
        gA[i] = A  + (size_t)(row0 + r + 32 * i) * lda + c * 8;
        gB[i] = Bm + (size_t)(col0 + r + 32 * i) * ldb + c * 8;
        lA[0][i] = lds_off(&As[0][(r + 32 * i) * LS + c * 8]);
        lA[1][i] = lds_off(&As[1][(r + 32 * i) * LS + c * 8]);
        lB[0][i] = lds_off(&Bs[0][(r + 32 * i) * LS + c * 8]);
        lB[1][i] = lds_off(&Bs[1][(r + 32 * i) * LS + c * 8]);
    }

    v8f acc[2][4] = {};

    // prologue: stage tile 0 into buffer 0
#pragma unroll
    for (int i = 0; i < 4; ++i) {
        async_copy_b128(lA[0][i], gA[i]);
        async_copy_b128(lB[0][i], gB[i]);
    }

    const int nk = K / KT;
    for (int t = 0; t < nk; ++t) {
        const int buf = t & 1;
        __syncthreads();                        // all waves done reading buf^1
        if (t + 1 < nk) {
            const int k1 = (t + 1) * KT;
#pragma unroll
            for (int i = 0; i < 4; ++i) {       // prefetch next tile into buf^1
                async_copy_b128(lA[buf ^ 1][i], gA[i] + k1);
                async_copy_b128(lB[buf ^ 1][i], gB[i] + k1);
            }
            __builtin_prefetch(gA[0] + k1 + KT, 0, 1);  // global_prefetch_b8
            __builtin_prefetch(gB[0] + k1 + KT, 0, 1);
            wait_async_le<8>();                 // older group (this buf) done
        } else {
            wait_async_le<0>();
        }
        __syncthreads();                        // buf visible to all waves

        const uint16_t* fA0 = &As[buf][(wm * 32 + l16)      * LS];
        const uint16_t* fA1 = &As[buf][(wm * 32 + 16 + l16) * LS];
        const uint16_t* fB0 = &Bs[buf][(wn * 64 + l16)      * LS];
#pragma unroll
        for (int kk = 0; kk < KT; kk += 32) {
            FragBF a0, a1, b;
            load_fragA(a0, fA0 + kk, h);
            load_fragA(a1, fA1 + kk, h);
#pragma unroll
            for (int j = 0; j < 4; ++j) {
                load_fragB(b, fB0 + j * 16 * LS + kk, h);
                acc[0][j] = __builtin_amdgcn_wmma_f32_16x16x32_bf16(
                    false, a0.v, false, b.v, (short)0, acc[0][j], false, false);
                acc[1][j] = __builtin_amdgcn_wmma_f32_16x16x32_bf16(
                    false, a1.v, false, b.v, (short)0, acc[1][j], false, false);
            }
        }
    }

    // Store. C/D layout: VGPR v <-> row (v + 8*half), lane%16 <-> col.
#pragma unroll
    for (int ti = 0; ti < 2; ++ti) {
#pragma unroll
        for (int tj = 0; tj < 4; ++tj) {
            AccF u; u.v = acc[ti][tj];
            const int col = col0 + wn * 64 + tj * 16 + l16;
#pragma unroll
            for (int v = 0; v < 8; ++v) {
                const int row = row0 + wm * 32 + ti * 16 + 8 * h + v;
                float val = u.f[v];
                if (EPI == 1) {
                    val += bias[col];
                    val = val > 0.0f ? val : 0.0f;
                    ((float*)Cv)[(size_t)row * ldc + col] = val;
                } else {
                    ((uint16_t*)Cv)[(size_t)row * ldc + col] = f2bf(val);
                }
            }
        }
    }
}

// --------------------------- row softmax (in place, bf16) ------------------

__global__ void softmax_rows(uint16_t* __restrict__ S, int ncols) {
    __shared__ float red[256];
    const int tid = threadIdx.x;
    uint16_t* row = S + (size_t)blockIdx.x * ncols;

    float m = -3.0e38f;
    for (int i = tid; i < ncols; i += 256) m = fmaxf(m, bf2f(row[i]));
    red[tid] = m; __syncthreads();
    for (int s = 128; s > 0; s >>= 1) {
        if (tid < s) red[tid] = fmaxf(red[tid], red[tid + s]);
        __syncthreads();
    }
    m = red[0]; __syncthreads();

    float sum = 0.0f;
    for (int i = tid; i < ncols; i += 256) sum += __expf(bf2f(row[i]) - m);
    red[tid] = sum; __syncthreads();
    for (int s = 128; s > 0; s >>= 1) {
        if (tid < s) red[tid] += red[tid + s];
        __syncthreads();
    }
    const float inv = 1.0f / red[0];

    for (int i = tid; i < ncols; i += 256)
        row[i] = f2bf(__expf(bf2f(row[i]) - m) * inv);
}

// --------------------------- launch ----------------------------------------

extern "C" void kernel_launch(void* const* d_in, const int* in_sizes, int n_in,
                              void* d_out, int out_size, void* d_ws, size_t ws_size,
                              hipStream_t stream) {
    (void)in_sizes; (void)n_in; (void)out_size; (void)ws_size;
    const int n = 4096, d = 1024;
    const float* Y  = (const float*)d_in[0];
    const float* W  = (const float*)d_in[1];
    const float* Wf = (const float*)d_in[2];
    const float* B  = (const float*)d_in[3];

    char* ws = (char*)d_ws;
    size_t off = 0;
    auto alloc = [&](size_t bytes) {
        void* p = ws + off;
        off += (bytes + 255) & ~(size_t)255;
        return p;
    };
    uint16_t* Ybf = (uint16_t*)alloc((size_t)n * d * 2);   // Y in bf16
    uint16_t* Ytb = (uint16_t*)alloc((size_t)n * d * 2);   // Y^T in bf16
    uint16_t* Wbf = (uint16_t*)alloc((size_t)d * d * 2);   // W in bf16
    uint16_t* Wfe = (uint16_t*)alloc((size_t)d * d * 2);   // -0.5(Wf+Wf^T), symmetric
    uint16_t* Qbf = (uint16_t*)alloc((size_t)n * d * 2);   // Q = Y W^T
    uint16_t* Sbf = (uint16_t*)alloc((size_t)n * n * 2);   // scores -> softmax in place
    uint16_t* Zbf = (uint16_t*)alloc((size_t)n * d * 2);   // Z = A Y

    prep_y <<<(n * d + 255) / 256, 256, 0, stream>>>(Y, Ybf, Ytb, n, d);
    prep_w <<<(d * d + 255) / 256, 256, 0, stream>>>(W, Wbf, d * d);
    prep_wf<<<(d * d + 255) / 256, 256, 0, stream>>>(Wf, Wfe, d);

    // Q = Ybf @ Wbf^T            (M=n, N=d, K=d)
    gemm_nt_wmma<0><<<dim3(d / 128, n / 128), 256, 0, stream>>>(Ybf, d, Wbf, d, Qbf, d, d, nullptr);
    // S = Qbf @ Qbf^T            (M=n, N=n, K=d)
    gemm_nt_wmma<0><<<dim3(n / 128, n / 128), 256, 0, stream>>>(Qbf, d, Qbf, d, Sbf, n, d, nullptr);
    // A = softmax(S) rows, in place
    softmax_rows<<<n, 256, 0, stream>>>(Sbf, n);
    // Z = A @ Y = A @ (Y^T)^T    (M=n, N=d, K=n)
    gemm_nt_wmma<0><<<dim3(d / 128, n / 128), 256, 0, stream>>>(Sbf, n, Ytb, n, Zbf, d, n, nullptr);
    // out = relu(Z @ Wfe + B)    (Wfe symmetric -> NT access is exact)
    gemm_nt_wmma<1><<<dim3(d / 128, n / 128), 256, 0, stream>>>(Zbf, d, Wfe, d, d_out, d, d, B);
}
